// RWKV7_Block_47425028882493
// MI455X (gfx1250) — matrix operand order repaired and checked by reading, not compiled, not run on hardware
//
#include <hip/hip_runtime.h>

// ---------------------------------------------------------------------------
// Types for gfx1250 WMMA (wave32).
// ---------------------------------------------------------------------------
typedef __attribute__((ext_vector_type(16))) __bf16 bf16x16;
typedef __attribute__((ext_vector_type(8)))  __bf16 bf16x8;
typedef __attribute__((ext_vector_type(8)))  float  f32x8;

#define DMODEL 1024
#define TLEN   2048
#define BATCH  8
#define MROWS  16384                 // B*T
#define XTOT   16777216              // MROWS*DMODEL

// ---------------------------------------------------------------------------
// WMMA helpers
// ---------------------------------------------------------------------------
static __device__ __forceinline__ f32x8 wmma_bf16f32(bf16x16 a, bf16x16 b, f32x8 c) {
    // (neg_a, A, neg_b, B, c_mod, C, reuse_a, reuse_b)
    return __builtin_amdgcn_wmma_f32_16x16x32_bf16(false, a, false, b, (short)0, c, false, false);
}

union FragU { bf16x16 v; bf16x8 h[2]; };

// A is [M, K=1024] row-major bf16.  16x32 fragment, contiguous k order (same as load_b_frag):
// lane L: row m = L&15, kb = 16*(L>>4); elems 0..15 = K kb..kb+15.
static __device__ __forceinline__ bf16x16 load_a_frag(const __bf16* A, int row0, int lane, int k0) {
    int m  = row0 + (lane & 15);
    int kb = k0 + ((lane >> 4) << 4);
    const __bf16* p = A + (size_t)m * DMODEL + kb;
    FragU f;
    f.h[0] = *reinterpret_cast<const bf16x8*>(p);
    f.h[1] = *reinterpret_cast<const bf16x8*>(p + 8);
    return f.v;
}

// BT is W^T, [N, K=1024] row-major bf16.  32x16 B fragment:
// lane L: col n = L&15, 16 contiguous K values starting at k0 + 16*(L>>4).
static __device__ __forceinline__ bf16x16 load_b_frag(const __bf16* BT, int col0, int lane, int k0) {
    int n  = col0 + (lane & 15);
    int kb = k0 + ((lane >> 4) << 4);
    return *reinterpret_cast<const bf16x16*>(BT + (size_t)n * DMODEL + kb);
}

// Fast sigmoid: denominator is in (1, +inf) but well-conditioned; v_rcp_f32
// (~1 ulp) avoids the IEEE div_scale/newton sequence in hot epilogues.
static __device__ __forceinline__ float sigmf(float x) {
    return __builtin_amdgcn_rcpf(1.0f + __expf(-x));
}

#define ZERO8 {0,0,0,0,0,0,0,0}

// ---------------------------------------------------------------------------
// Weight convert + transpose: W[K,N] f32 -> WT[N,K] bf16
// ---------------------------------------------------------------------------
__global__ __launch_bounds__(256) void wcvt_kernel(const float* __restrict__ W, __bf16* __restrict__ WT) {
    int idx = blockIdx.x * 256 + threadIdx.x;      // < 1048576
    int r = idx >> 10, c = idx & 1023;
    WT[(size_t)c * DMODEL + r] = (__bf16)W[idx];
}

// ---------------------------------------------------------------------------
// LN1 + token-shift mix: xm = 0.5*(LN(x[t]) + LN(x[t-1])), bf16 out
// ---------------------------------------------------------------------------
__global__ __launch_bounds__(256) void ln1mix_kernel(const float* __restrict__ X,
                                                     const float* __restrict__ w,
                                                     const float* __restrict__ bi,
                                                     __bf16* __restrict__ out) {
    int row = blockIdx.x;
    int t   = row & (TLEN - 1);
    int tid = threadIdx.x;
    const float* xc = X + (size_t)row * DMODEL;
    const float* xp = xc - DMODEL;                 // valid only when t>0
    float vc[4], vp[4];
    float s1 = 0.f, q1 = 0.f, s0 = 0.f, q0 = 0.f;
#pragma unroll
    for (int i = 0; i < 4; i++) {
        int c = tid + (i << 8);
        float a = xc[c]; vc[i] = a; s1 += a; q1 += a * a;
        float p = (t > 0) ? xp[c] : 0.0f; vp[i] = p; s0 += p; q0 += p * p;
    }
    __shared__ float r1[256], r2[256], r3[256], r4[256];
    r1[tid] = s1; r2[tid] = q1; r3[tid] = s0; r4[tid] = q0;
    __syncthreads();
    for (int off = 128; off > 0; off >>= 1) {
        if (tid < off) { r1[tid] += r1[tid+off]; r2[tid] += r2[tid+off];
                         r3[tid] += r3[tid+off]; r4[tid] += r4[tid+off]; }
        __syncthreads();
    }
    const float invd = 1.0f / (float)DMODEL;
    float m1 = r1[0] * invd, v1 = r2[0] * invd - m1 * m1;
    float i1 = rsqrtf(v1 + 1e-5f);
    float m0 = r3[0] * invd, v0 = r4[0] * invd - m0 * m0;
    float i0 = rsqrtf(v0 + 1e-5f);
#pragma unroll
    for (int i = 0; i < 4; i++) {
        int c = tid + (i << 8);
        float n1 = (vc[i] - m1) * i1 * w[c] + bi[c];
        float n0 = (t > 0) ? ((vp[i] - m0) * i0 * w[c] + bi[c]) : 0.0f;
        out[(size_t)row * DMODEL + c] = (__bf16)(0.5f * (n1 + n0));
    }
}

// ---------------------------------------------------------------------------
// Fused dual GEMM, 4 M-tiles/wave: rg = sigmoid(A@Wr) * sigmoid(A@Wg)
// grid (64, 32), block 256
// ---------------------------------------------------------------------------
__global__ __launch_bounds__(256) void gemm_rg_kernel(const __bf16* __restrict__ A,
                                                      const __bf16* __restrict__ BTr,
                                                      const __bf16* __restrict__ BTg,
                                                      float* __restrict__ out) {
    int lane = threadIdx.x & 31, wave = threadIdx.x >> 5;
    int col0 = blockIdx.x << 4;
    int row0 = ((blockIdx.y << 3) + wave) << 6;        // 64 rows per wave
    f32x8 ar[4] = {ZERO8, ZERO8, ZERO8, ZERO8};
    f32x8 ag[4] = {ZERO8, ZERO8, ZERO8, ZERO8};
    for (int k0 = 0; k0 < DMODEL; k0 += 32) {
        bf16x16 af[4];
#pragma unroll
        for (int m = 0; m < 4; m++) af[m] = load_a_frag(A, row0 + (m << 4), lane, k0);
        bf16x16 br = load_b_frag(BTr, col0, lane, k0);
#pragma unroll
        for (int m = 0; m < 4; m++) ar[m] = wmma_bf16f32(af[m], br, ar[m]);
        bf16x16 bg = load_b_frag(BTg, col0, lane, k0);
#pragma unroll
        for (int m = 0; m < 4; m++) ag[m] = wmma_bf16f32(af[m], bg, ag[m]);
    }
    int n = col0 + (lane & 15);
#pragma unroll
    for (int m = 0; m < 4; m++) {
        int m0 = row0 + (m << 4) + ((lane >> 4) << 3);
#pragma unroll
        for (int r = 0; r < 8; r++)
            out[(size_t)(m0 + r) * DMODEL + n] = sigmf(ar[m][r]) * sigmf(ag[m][r]);
    }
}

// ---------------------------------------------------------------------------
// Fused quad GEMM, 2 M-tiles/wave: u = (A@Wk)*(A@Wv) + (A@Wa)*(A@Wb)
// grid (64, 64), block 256
// ---------------------------------------------------------------------------
__global__ __launch_bounds__(256) void gemm_u_kernel(const __bf16* __restrict__ A,
                                                     const __bf16* __restrict__ BTk,
                                                     const __bf16* __restrict__ BTv,
                                                     const __bf16* __restrict__ BTa,
                                                     const __bf16* __restrict__ BTb,
                                                     float* __restrict__ out) {
    int lane = threadIdx.x & 31, wave = threadIdx.x >> 5;
    int col0 = blockIdx.x << 4;
    int row0 = ((blockIdx.y << 3) + wave) << 5;        // 32 rows per wave
    f32x8 ck[2] = {ZERO8, ZERO8}, cv[2] = {ZERO8, ZERO8};
    f32x8 ca[2] = {ZERO8, ZERO8}, cb[2] = {ZERO8, ZERO8};
    for (int k0 = 0; k0 < DMODEL; k0 += 32) {
        bf16x16 af[2];
#pragma unroll
        for (int m = 0; m < 2; m++) af[m] = load_a_frag(A, row0 + (m << 4), lane, k0);
        bf16x16 bk = load_b_frag(BTk, col0, lane, k0);
#pragma unroll
        for (int m = 0; m < 2; m++) ck[m] = wmma_bf16f32(af[m], bk, ck[m]);
        bf16x16 bv = load_b_frag(BTv, col0, lane, k0);
#pragma unroll
        for (int m = 0; m < 2; m++) cv[m] = wmma_bf16f32(af[m], bv, cv[m]);
        bf16x16 ba = load_b_frag(BTa, col0, lane, k0);
#pragma unroll
        for (int m = 0; m < 2; m++) ca[m] = wmma_bf16f32(af[m], ba, ca[m]);
        bf16x16 bb = load_b_frag(BTb, col0, lane, k0);
#pragma unroll
        for (int m = 0; m < 2; m++) cb[m] = wmma_bf16f32(af[m], bb, cb[m]);
    }
    int n = col0 + (lane & 15);
#pragma unroll
    for (int m = 0; m < 2; m++) {
        int m0 = row0 + (m << 4) + ((lane >> 4) << 3);
#pragma unroll
        for (int r = 0; r < 8; r++)
            out[(size_t)(m0 + r) * DMODEL + n] = ck[m][r] * cv[m][r] + ca[m][r] * cb[m][r];
    }
}

// ---------------------------------------------------------------------------
// GEMM with residual add, 4 M-tiles x 2 N-tiles/wave: out = res + A@W
// (res may alias out).  grid (32, 32), block 256
// ---------------------------------------------------------------------------
__global__ __launch_bounds__(256) void gemm_resadd_kernel(const __bf16* __restrict__ A,
                                                          const __bf16* __restrict__ BT,
                                                          const float* __restrict__ res,
                                                          float* __restrict__ out) {
    int lane = threadIdx.x & 31, wave = threadIdx.x >> 5;
    int col0 = blockIdx.x << 5;                        // 32 cols per wave
    int row0 = ((blockIdx.y << 3) + wave) << 6;        // 64 rows per wave
    f32x8 acc[4][2] = {{ZERO8, ZERO8}, {ZERO8, ZERO8}, {ZERO8, ZERO8}, {ZERO8, ZERO8}};
    for (int k0 = 0; k0 < DMODEL; k0 += 32) {
        bf16x16 af[4];
#pragma unroll
        for (int m = 0; m < 4; m++) af[m] = load_a_frag(A, row0 + (m << 4), lane, k0);
#pragma unroll
        for (int nn = 0; nn < 2; nn++) {
            bf16x16 b = load_b_frag(BT, col0 + (nn << 4), lane, k0);
#pragma unroll
            for (int m = 0; m < 4; m++) acc[m][nn] = wmma_bf16f32(af[m], b, acc[m][nn]);
        }
    }
    int nbase = col0 + (lane & 15);
#pragma unroll
    for (int m = 0; m < 4; m++) {
        int m0 = row0 + (m << 4) + ((lane >> 4) << 3);
#pragma unroll
        for (int nn = 0; nn < 2; nn++) {
            int n = nbase + (nn << 4);
#pragma unroll
            for (int r = 0; r < 8; r++) {
                size_t idx = (size_t)(m0 + r) * DMODEL + n;
                out[idx] = res[idx] + acc[m][nn][r];
            }
        }
    }
}

// ---------------------------------------------------------------------------
// GEMM + bias + activation, 4 M-tiles x 2 N-tiles/wave.
// MODE 0: bf16(sq(relu(acc+b)))  MODE 1: f32 acc+b  MODE 2: f32 sigmoid(acc+b)
// grid (32, 32), block 256
// ---------------------------------------------------------------------------
template <int MODE>
__global__ __launch_bounds__(256) void gemm_bias_kernel(const __bf16* __restrict__ A,
                                                        const __bf16* __restrict__ BT,
                                                        const float* __restrict__ bias,
                                                        float* __restrict__ outF,
                                                        __bf16* __restrict__ outH) {
    int lane = threadIdx.x & 31, wave = threadIdx.x >> 5;
    int col0 = blockIdx.x << 5;
    int row0 = ((blockIdx.y << 3) + wave) << 6;
    f32x8 acc[4][2] = {{ZERO8, ZERO8}, {ZERO8, ZERO8}, {ZERO8, ZERO8}, {ZERO8, ZERO8}};
    for (int k0 = 0; k0 < DMODEL; k0 += 32) {
        bf16x16 af[4];
#pragma unroll
        for (int m = 0; m < 4; m++) af[m] = load_a_frag(A, row0 + (m << 4), lane, k0);
#pragma unroll
        for (int nn = 0; nn < 2; nn++) {
            bf16x16 b = load_b_frag(BT, col0 + (nn << 4), lane, k0);
#pragma unroll
            for (int m = 0; m < 4; m++) acc[m][nn] = wmma_bf16f32(af[m], b, acc[m][nn]);
        }
    }
    int nbase = col0 + (lane & 15);
    float bs[2] = {bias[nbase], bias[nbase + 16]};
#pragma unroll
    for (int m = 0; m < 4; m++) {
        int m0 = row0 + (m << 4) + ((lane >> 4) << 3);
#pragma unroll
        for (int nn = 0; nn < 2; nn++) {
            int n = nbase + (nn << 4);
#pragma unroll
            for (int r = 0; r < 8; r++) {
                size_t idx = (size_t)(m0 + r) * DMODEL + n;
                float v = acc[m][nn][r] + bs[nn];
                if (MODE == 0) { float t = fmaxf(v, 0.0f); outH[idx] = (__bf16)(t * t); }
                if (MODE == 1) { outF[idx] = v; }
                if (MODE == 2) { outF[idx] = sigmf(v); }
            }
        }
    }
}

// ---------------------------------------------------------------------------
// Per-channel recurrence: s = s*w + u, in place (u -> states), emit final state
// ---------------------------------------------------------------------------
__global__ __launch_bounds__(256) void scan_kernel(float* __restrict__ U,
                                                   const float* __restrict__ st0,
                                                   const float* __restrict__ decay,
                                                   float* __restrict__ fstate) {
    int tid = blockIdx.x * 256 + threadIdx.x;      // 0..8191
    int b = tid >> 10, d = tid & 1023;
    float w = __expf(-__expf(decay[d]));
    float s = st0[b * DMODEL + d];
    size_t base = ((size_t)b << 21) + d;
    for (int t = 0; t < TLEN; t++) {
        size_t idx = base + ((size_t)t << 10);
        s = s * w + U[idx];
        U[idx] = s;
    }
    fstate[b * DMODEL + d] = s;
}

// ---------------------------------------------------------------------------
// GroupNorm(32-wide groups) * rg  -> bf16
// ---------------------------------------------------------------------------
__global__ __launch_bounds__(256) void gnmix_kernel(const float* __restrict__ S,
                                                    const float* __restrict__ RG,
                                                    const float* __restrict__ gw,
                                                    const float* __restrict__ gb,
                                                    __bf16* __restrict__ out) {
    int row = blockIdx.x, tid = threadIdx.x;
    __shared__ float sh[DMODEL];
    __shared__ float gm[32], gi[32];
    const float* sr = S + (size_t)row * DMODEL;
#pragma unroll
    for (int i = 0; i < 4; i++) sh[tid + (i << 8)] = sr[tid + (i << 8)];
    __syncthreads();
    if (tid < 32) {
        float s = 0.f, q = 0.f;
#pragma unroll
        for (int j = 0; j < 32; j++) { float v = sh[tid * 32 + j]; s += v; q += v * v; }
        float m = s * (1.0f / 32.0f);
        float var = q * (1.0f / 32.0f) - m * m;
        gm[tid] = m; gi[tid] = rsqrtf(var + 1e-5f);
    }
    __syncthreads();
#pragma unroll
    for (int i = 0; i < 4; i++) {
        int c = tid + (i << 8); int g = c >> 5;
        float v = (sh[c] - gm[g]) * gi[g] * gw[c] + gb[c];
        out[(size_t)row * DMODEL + c] = (__bf16)(v * RG[(size_t)row * DMODEL + c]);
    }
}

// ---------------------------------------------------------------------------
// Plain LayerNorm -> bf16 (for LN2)
// ---------------------------------------------------------------------------
__global__ __launch_bounds__(256) void ln_kernel(const float* __restrict__ X,
                                                 const float* __restrict__ w,
                                                 const float* __restrict__ bi,
                                                 __bf16* __restrict__ out) {
    int row = blockIdx.x, tid = threadIdx.x;
    const float* xr = X + (size_t)row * DMODEL;
    float vals[4]; float s = 0.f, q = 0.f;
#pragma unroll
    for (int i = 0; i < 4; i++) { float v = xr[tid + (i << 8)]; vals[i] = v; s += v; q += v * v; }
    __shared__ float ss[256], sq[256];
    ss[tid] = s; sq[tid] = q;
    __syncthreads();
    for (int off = 128; off > 0; off >>= 1) {
        if (tid < off) { ss[tid] += ss[tid+off]; sq[tid] += sq[tid+off]; }
        __syncthreads();
    }
    const float invd = 1.0f / (float)DMODEL;
    float m = ss[0] * invd;
    float var = sq[0] * invd - m * m;
    float inv = rsqrtf(var + 1e-5f);
#pragma unroll
    for (int i = 0; i < 4; i++) {
        int c = tid + (i << 8);
        out[(size_t)row * DMODEL + c] = (__bf16)((vals[i] - m) * inv * w[c] + bi[c]);
    }
}

// ---------------------------------------------------------------------------
// GRN stage 1: gx[b,d] = sqrt(sum_t h^2)
// ---------------------------------------------------------------------------
__global__ __launch_bounds__(256) void grn_gx_kernel(const float* __restrict__ H,
                                                     float* __restrict__ gx) {
    int tid = blockIdx.x * 256 + threadIdx.x;      // 0..8191
    int b = tid >> 10, d = tid & 1023;
    size_t base = ((size_t)b << 21) + d;
    float acc = 0.f;
    for (int t = 0; t < TLEN; t++) { float v = H[base + ((size_t)t << 10)]; acc += v * v; }
    gx[b * DMODEL + d] = sqrtf(acc);
}

// ---------------------------------------------------------------------------
// GRN stage 2: nx[b,d] = gx / (mean_d gx + 1e-6)
// ---------------------------------------------------------------------------
__global__ __launch_bounds__(256) void grn_nx_kernel(const float* __restrict__ gx,
                                                     float* __restrict__ nx) {
    int b = blockIdx.x, tid = threadIdx.x;
    __shared__ float ss[256];
    float s = 0.f;
#pragma unroll
    for (int i = 0; i < 4; i++) s += gx[b * DMODEL + tid + (i << 8)];
    ss[tid] = s;
    __syncthreads();
    for (int off = 128; off > 0; off >>= 1) {
        if (tid < off) ss[tid] += ss[tid+off];
        __syncthreads();
    }
    float inv = 1.0f / (ss[0] * (1.0f / (float)DMODEL) + 1e-6f);
#pragma unroll
    for (int i = 0; i < 4; i++) {
        int c = tid + (i << 8);
        nx[b * DMODEL + c] = gx[b * DMODEL + c] * inv;
    }
}

// ---------------------------------------------------------------------------
// ffn_in = frsig * (gamma*(h*nx) + beta + h) -> bf16
// ---------------------------------------------------------------------------
__global__ __launch_bounds__(256) void ffnin_kernel(const float* __restrict__ H,
                                                    const float* __restrict__ FR,
                                                    const float* __restrict__ nx,
                                                    const float* __restrict__ gamma,
                                                    const float* __restrict__ beta,
                                                    __bf16* __restrict__ out) {
    size_t idx = (size_t)blockIdx.x * 256 + threadIdx.x;   // < 16777216
    int d = (int)(idx & 1023);
    int b = (int)(idx >> 21);
    float h = H[idx];
    float val = gamma[d] * (h * nx[b * DMODEL + d]) + beta[d] + h;
    out[idx] = (__bf16)(FR[idx] * val);
}

// ---------------------------------------------------------------------------
// Host launcher
// ---------------------------------------------------------------------------
extern "C" void kernel_launch(void* const* d_in, const int* in_sizes, int n_in,
                              void* d_out, int out_size, void* d_ws, size_t ws_size,
                              hipStream_t stream) {
    (void)in_sizes; (void)n_in; (void)out_size; (void)ws_size;

    const float* x         = (const float*)d_in[0];
    const float* state     = (const float*)d_in[1];
    const float* ln1_w     = (const float*)d_in[2];
    const float* ln1_b     = (const float*)d_in[3];
    const float* ln2_w     = (const float*)d_in[4];
    const float* ln2_b     = (const float*)d_in[5];
    const float* gn_w      = (const float*)d_in[6];
    const float* gn_b      = (const float*)d_in[7];
    const float* grn_gamma = (const float*)d_in[8];
    const float* grn_beta  = (const float*)d_in[9];
    const float* decay     = (const float*)d_in[10];
    const float* bfk       = (const float*)d_in[20];
    const float* bfv       = (const float*)d_in[22];
    const float* bfr       = (const float*)d_in[24];

    // Workspace layout (bytes):
    //   [0, 23068672)            : 11 transposed bf16 weights (2MB each)
    //   [23068672, +32MB)        : R1 bf16 [16384,1024]  (xm, later xx2)
    //   [56623104, +64MB)        : R2 f32  (u -> states, later h)
    //   [123731968, +64MB)       : R3 f32  (rg, later frsig)
    //   [190840832, +32MB)       : R4 bf16 (mix_in -> hk -> ffn_in)
    //   [224395264, +32KB)       : gx f32 [8,1024]
    //   [224428032, +32KB)       : nx f32 [8,1024]
    char* ws = (char*)d_ws;
    __bf16* WT0 = (__bf16*)ws;
    __bf16* R1h = (__bf16*)(ws + 23068672);
    float*  R2f = (float*)(ws + 56623104);
    float*  R3f = (float*)(ws + 123731968);
    __bf16* R4h = (__bf16*)(ws + 190840832);
    float*  gx  = (float*)(ws + 224395264);
    float*  nxp = (float*)(ws + 224428032);

    float* outx   = (float*)d_out;
    float* fstate = (float*)d_out + XTOT;

    // Weight order in WT region: Wr Wk Wv Wg Wa Wb Wout Wffnout Wfk Wfv Wfr
    const int wsrc[11] = {11, 12, 13, 14, 15, 16, 17, 18, 19, 21, 23};
    for (int i = 0; i < 11; i++)
        wcvt_kernel<<<4096, 256, 0, stream>>>((const float*)d_in[wsrc[i]],
                                              WT0 + (size_t)i * (DMODEL * DMODEL));
    enum { IWr = 0, IWk, IWv, IWg, IWa, IWb, IWout, IWffnout, IWfk, IWfv, IWfr };
    auto WT = [&](int i) { return WT0 + (size_t)i * (DMODEL * DMODEL); };

    dim3 g_rg(DMODEL / 16, MROWS / 512);   // (64, 32): 4 M-tiles/wave
    dim3 g_u(DMODEL / 16, MROWS / 256);    // (64, 64): 2 M-tiles/wave
    dim3 g_1w(DMODEL / 32, MROWS / 512);   // (32, 32): 4x2 tiles/wave

    // 1) LN1 + token-shift mix -> xm (bf16)
    ln1mix_kernel<<<MROWS, 256, 0, stream>>>(x, ln1_w, ln1_b, R1h);
    // 2) rg = sigmoid(xm@Wr) * sigmoid(xm@Wg)
    gemm_rg_kernel<<<g_rg, 256, 0, stream>>>(R1h, WT(IWr), WT(IWg), R3f);
    // 3) u = (xm@Wk)*(xm@Wv) + (xm@Wa)*(xm@Wb)
    gemm_u_kernel<<<g_u, 256, 0, stream>>>(R1h, WT(IWk), WT(IWv), WT(IWa), WT(IWb), R2f);
    // 4) time scan in place; write final state
    scan_kernel<<<32, 256, 0, stream>>>(R2f, state, decay, fstate);
    // 5) GroupNorm(states) * rg -> mix_in (bf16)
    gnmix_kernel<<<MROWS, 256, 0, stream>>>(R2f, R3f, gn_w, gn_b, R4h);
    // 6) x2 = x + mix_in @ Wout  -> d_out
    gemm_resadd_kernel<<<g_1w, 256, 0, stream>>>(R4h, WT(IWout), x, outx);
    // 7) LN2(x2) -> xx2 (bf16)
    ln_kernel<<<MROWS, 256, 0, stream>>>(outx, ln2_w, ln2_b, R1h);
    // 8) hk = sq(relu(xx2@Wfk + bfk)) (bf16)
    gemm_bias_kernel<0><<<g_1w, 256, 0, stream>>>(R1h, WT(IWfk), bfk, nullptr, R4h);
    // 9) h = hk@Wfv + bfv (f32)
    gemm_bias_kernel<1><<<g_1w, 256, 0, stream>>>(R4h, WT(IWfv), bfv, R2f, nullptr);
    // 10) frsig = sigmoid(xx2@Wfr + bfr) (f32)
    gemm_bias_kernel<2><<<g_1w, 256, 0, stream>>>(R1h, WT(IWfr), bfr, R3f, nullptr);
    // 11) GRN reductions
    grn_gx_kernel<<<32, 256, 0, stream>>>(R2f, gx);
    grn_nx_kernel<<<BATCH, 256, 0, stream>>>(gx, nxp);
    // 12) ffn_in = frsig * (gamma*(h*nx) + beta + h) (bf16)
    ffnin_kernel<<<XTOT / 256, 256, 0, stream>>>(R2f, R3f, nxp, grn_gamma, grn_beta, R4h);
    // 13) d_out += ffn_in @ Wffnout
    gemm_resadd_kernel<<<g_1w, 256, 0, stream>>>(R4h, WT(IWffnout), outx, outx);
}
